// TransformerEncoderLayer_19318762897507
// MI455X (gfx1250) — matrix operand order repaired and checked by reading, not compiled, not run on hardware
//
#include <hip/hip_runtime.h>
#include <hip/hip_bf16.h>

typedef _Float16 half_t;
typedef __attribute__((ext_vector_type(16))) _Float16 v16h;
typedef __attribute__((ext_vector_type(8)))  _Float16 v8h;
typedef __attribute__((ext_vector_type(8)))  float    v8f;

#define D_MODEL   1024
#define NUM_HEADS 16
#define HEAD_DIM  64
#define FF_DIM    4096
#define BATCH     4
#define SEQ       1024
#define BN        (BATCH * SEQ)
#define LN_EPS    1e-5f

// ---------------------------------------------------------------------------
// CDNA5 async global->LDS DMA (ASYNCcnt-tracked), used for GEMM tile staging.
// GVS mode: saddr = 64-bit uniform base (SGPR pair), vaddr = 32-bit per-lane
// byte offset, VDST = per-lane 32-bit LDS byte address.
// ---------------------------------------------------------------------------
typedef __attribute__((address_space(3))) const char lds_cchar;
__device__ __forceinline__ unsigned lds_offset(const void* p) {
    return (unsigned)(unsigned long long)(lds_cchar*)p;
}

#define ASYNC_LOAD_B128(saddr, voff, ldsoff)                      \
    asm volatile("global_load_async_to_lds_b128 %0, %1, %2"       \
                 :: "v"(ldsoff), "v"(voff), "s"(saddr) : "memory")

#define WAIT_ASYNCCNT(n) asm volatile("s_wait_asynccnt " #n ::: "memory")

// ---------------------------------------------------------------------------
// WMMA fragment loaders (wave32, V_WMMA_F32_16X16X32_F16 layouts per CDNA5 ISA)
// A (16x32, MxK):  lane L: m = L%16; K = {0..7,16..23} (L<16) / {8..15,24..31}
// B (32x16, KxN):  lane L: n = L%16; K = {0..7,16..23} (L<16) / {8..15,24..31}
// Both loaders read from LDS tiles stored row-major with contiguous K.
// ---------------------------------------------------------------------------
__device__ __forceinline__ v16h load_frag_a(const half_t* base, int lane, int stride) {
    int m  = lane & 15;
    int k0 = (lane & 16) ? 8 : 0;
    const half_t* p = base + m * stride + k0;
    union { v16h v; v8h h[2]; } u;
    u.h[0] = *(const v8h*)(p);
    u.h[1] = *(const v8h*)(p + 16);
    return u.v;
}

__device__ __forceinline__ v16h load_frag_b(const half_t* base, int lane, int stride) {
    int n  = lane & 15;
    int k0 = (lane & 16) ? 8 : 0;
    const half_t* p = base + n * stride + k0;
    union { v16h v; v8h h[2]; } u;
    u.h[0] = *(const v8h*)(p);
    u.h[1] = *(const v8h*)(p + 16);
    return u.v;
}

// ---------------------------------------------------------------------------
// Elementwise prep: qk16 = f16(input + pos); in16 = f16(input)
// ---------------------------------------------------------------------------
__global__ void prep_inputs(const float* __restrict__ in, const float* __restrict__ pos,
                            half_t* __restrict__ qk16, half_t* __restrict__ in16, int n) {
    int i = blockIdx.x * blockDim.x + threadIdx.x;
    if (i < n) {
        float a = in[i];
        qk16[i] = (half_t)(a + pos[i]);
        in16[i] = (half_t)a;
    }
}

// W is [K][N] row-major f32 -> WT is [N][K] row-major f16 (K contiguous)
__global__ void convert_transpose(const float* __restrict__ W, half_t* __restrict__ WT,
                                  int K, int N) {
    int i = blockIdx.x * blockDim.x + threadIdx.x;
    if (i < K * N) {
        int k = i / N, n = i % N;
        WT[(size_t)n * K + k] = (half_t)W[i];
    }
}

// ---------------------------------------------------------------------------
// Tiled WMMA GEMM:  C[M][N] = A[M][K] * BT[N][K]^T  (+bias)(+relu)
// 128 threads = 4 waves, 64x64 output tile, K-chunk 32.
// Tiles staged with double-buffered async global->LDS DMA:
//   issue chunk c+1 -> s_wait_asynccnt 4 (chunk c landed, in-order per wave)
//   -> barrier -> WMMA on chunk c. 4 async b128 per thread per chunk.
// ---------------------------------------------------------------------------
template<bool BIAS, bool RELU, bool O16, bool O32>
__global__ __launch_bounds__(128)
void gemm_wmma(const half_t* __restrict__ A, const half_t* __restrict__ BT,
               const float* __restrict__ bias,
               half_t* __restrict__ out16, float* __restrict__ out32,
               int M, int N, int K) {
    constexpr int AST = 40;  // 80B row stride: 16B-aligned vectors, staggered banks
    __shared__ half_t As[2][64 * AST] __attribute__((aligned(16)));
    __shared__ half_t Bs[2][64 * AST] __attribute__((aligned(16)));

    const int tid  = threadIdx.x;
    const int wave = tid >> 5;
    const int lane = tid & 31;
    const int m0 = blockIdx.y * 64;
    const int n0 = blockIdx.x * 64;

    // This thread's two 16B transfer slots within each 64x32 tile.
    const int row = tid >> 1;                 // 0..63
    const int cv0 = (tid & 1) ? 2 : 0;        // vector column 0..3
    const int cv1 = cv0 + 1;

    auto issue_tiles = [&](int buf, int kc) {
        unsigned arow = ((unsigned)(m0 + row) * (unsigned)K + (unsigned)kc) * 2u;
        unsigned brow = ((unsigned)(n0 + row) * (unsigned)K + (unsigned)kc) * 2u;
        unsigned la0 = lds_offset(&As[buf][row * AST + cv0 * 8]);
        unsigned la1 = lds_offset(&As[buf][row * AST + cv1 * 8]);
        unsigned lb0 = lds_offset(&Bs[buf][row * AST + cv0 * 8]);
        unsigned lb1 = lds_offset(&Bs[buf][row * AST + cv1 * 8]);
        ASYNC_LOAD_B128(A,  arow + cv0 * 16u, la0);
        ASYNC_LOAD_B128(A,  arow + cv1 * 16u, la1);
        ASYNC_LOAD_B128(BT, brow + cv0 * 16u, lb0);
        ASYNC_LOAD_B128(BT, brow + cv1 * 16u, lb1);
    };

    v8f acc[4] = {};
    const int nchunks = K >> 5;

    issue_tiles(0, 0);
    for (int c = 0; c < nchunks; ++c) {
        const int cur = c & 1;
        __syncthreads();                 // all waves done reading buf[cur^1]
        if (c + 1 < nchunks) {
            issue_tiles(cur ^ 1, (c + 1) << 5);
            WAIT_ASYNCCNT(4);            // chunk c's 4 DMAs done (in-order)
        } else {
            WAIT_ASYNCCNT(0);
        }
        __syncthreads();                 // chunk c visible to all waves

        v16h a = load_frag_a(&As[cur][16 * wave * AST], lane, AST);
        #pragma unroll
        for (int j = 0; j < 4; ++j) {
            v16h b = load_frag_b(&Bs[cur][16 * j * AST], lane, AST);
            acc[j] = __builtin_amdgcn_wmma_f32_16x16x32_f16(
                false, a, false, b, (short)0, acc[j], false, false);
        }
    }

    #pragma unroll
    for (int j = 0; j < 4; ++j) {
        int n = n0 + 16 * j + (lane & 15);
        float bv = BIAS ? bias[n] : 0.0f;
        #pragma unroll
        for (int r = 0; r < 8; ++r) {
            int m = m0 + 16 * wave + r + ((lane & 16) ? 8 : 0);
            float val = acc[j][r] + bv;
            if (RELU) val = fmaxf(val, 0.0f);
            if (O16) out16[(size_t)m * N + n] = (half_t)val;
            if (O32) out32[(size_t)m * N + n] = val;
        }
    }
}

// ---------------------------------------------------------------------------
// Flash attention: grid (SEQ/64, B*H), 128 threads (4 waves, 16 q-rows each).
// S = (Q K^T) * sqrt(head_dim)   [faithful quirk]; online softmax; O += P V.
// ---------------------------------------------------------------------------
__global__ __launch_bounds__(128)
void attn_flash(const half_t* __restrict__ Q, const half_t* __restrict__ Kd,
                const half_t* __restrict__ V, float* __restrict__ O) {
    constexpr int ST = 72;  // 144B row stride (16B multiple)
    __shared__ half_t Qs[64 * ST] __attribute__((aligned(16)));
    __shared__ half_t Ks[64 * ST] __attribute__((aligned(16)));
    __shared__ half_t Vt[64 * ST] __attribute__((aligned(16)));   // Vt[d][key]
    __shared__ half_t Ps[4 * 16 * 64] __attribute__((aligned(16)));  // per-wave P

    const int tid  = threadIdx.x;
    const int wave = tid >> 5;
    const int lane = tid & 31;
    const int bh = blockIdx.y;
    const int b  = bh >> 4;
    const int h  = bh & 15;
    const int q0 = blockIdx.x * 64;
    const size_t rowbase = (size_t)b * SEQ;
    const int c0 = h * HEAD_DIM;

    // Q tile resident for the whole kernel
    #pragma unroll
    for (int i = 0; i < 4; ++i) {
        int v = tid * 4 + i;
        int row = v >> 3, cv = v & 7;
        *(v8h*)(Qs + row * ST + cv * 8) =
            *(const v8h*)(Q + (rowbase + q0 + row) * D_MODEL + c0 + cv * 8);
    }

    v8f oacc[4] = {};
    float mrun[8], lrun[8];
    #pragma unroll
    for (int r = 0; r < 8; ++r) { mrun[r] = -3.0e38f; lrun[r] = 0.0f; }

    for (int k0 = 0; k0 < SEQ; k0 += 64) {
        __syncthreads();
        #pragma unroll
        for (int i = 0; i < 4; ++i) {
            int v = tid * 4 + i;
            int row = v >> 3, cv = v & 7;
            *(v8h*)(Ks + row * ST + cv * 8) =
                *(const v8h*)(Kd + (rowbase + k0 + row) * D_MODEL + c0 + cv * 8);
            v8h vv = *(const v8h*)(V + (rowbase + k0 + row) * D_MODEL + c0 + cv * 8);
            #pragma unroll
            for (int e = 0; e < 8; ++e) Vt[(cv * 8 + e) * ST + row] = vv[e];
        }
        __syncthreads();

        // S = Q K^T  (head_dim 64 = two K-chunks of 32)
        v8f sacc[4] = {};
        v16h a0 = load_frag_a(Qs + 16 * wave * ST, lane, ST);
        v16h a1 = load_frag_a(Qs + 16 * wave * ST + 32, lane, ST);
        #pragma unroll
        for (int j = 0; j < 4; ++j) {
            v16h b0 = load_frag_b(Ks + 16 * j * ST, lane, ST);
            v16h b1 = load_frag_b(Ks + 16 * j * ST + 32, lane, ST);
            sacc[j] = __builtin_amdgcn_wmma_f32_16x16x32_f16(
                false, a0, false, b0, (short)0, sacc[j], false, false);
            sacc[j] = __builtin_amdgcn_wmma_f32_16x16x32_f16(
                false, a1, false, b1, (short)0, sacc[j], false, false);
        }

        const float scale = 8.0f;  // sqrt(HEAD_DIM): reference divides by d^-0.5
        half_t* PsW = Ps + wave * 16 * 64;
        #pragma unroll
        for (int r = 0; r < 8; ++r) {
            float rmax = -3.0e38f;
            #pragma unroll
            for (int j = 0; j < 4; ++j) {
                sacc[j][r] *= scale;
                rmax = fmaxf(rmax, sacc[j][r]);
            }
            #pragma unroll
            for (int off = 8; off >= 1; off >>= 1)
                rmax = fmaxf(rmax, __shfl_xor(rmax, off, 32));
            float mn = fmaxf(mrun[r], rmax);
            float alpha = __expf(mrun[r] - mn);
            mrun[r] = mn;
            float rsum = 0.0f;
            int m = r + ((lane & 16) ? 8 : 0);
            #pragma unroll
            for (int j = 0; j < 4; ++j) {
                float p = __expf(sacc[j][r] - mn);
                rsum += p;
                PsW[m * 64 + 16 * j + (lane & 15)] = (half_t)p;  // C-layout -> A-layout via LDS
            }
            #pragma unroll
            for (int off = 8; off >= 1; off >>= 1)
                rsum += __shfl_xor(rsum, off, 32);
            lrun[r] = lrun[r] * alpha + rsum;
            #pragma unroll
            for (int j = 0; j < 4; ++j) oacc[j][r] *= alpha;
        }

        // O += P * V   (keys are the K-dim: two chunks of 32)
        v16h pa0 = load_frag_a(PsW, lane, 64);
        v16h pa1 = load_frag_a(PsW + 32, lane, 64);
        #pragma unroll
        for (int j = 0; j < 4; ++j) {
            v16h vb0 = load_frag_b(Vt + 16 * j * ST, lane, ST);
            v16h vb1 = load_frag_b(Vt + 16 * j * ST + 32, lane, ST);
            oacc[j] = __builtin_amdgcn_wmma_f32_16x16x32_f16(
                false, pa0, false, vb0, (short)0, oacc[j], false, false);
            oacc[j] = __builtin_amdgcn_wmma_f32_16x16x32_f16(
                false, pa1, false, vb1, (short)0, oacc[j], false, false);
        }
    }

    #pragma unroll
    for (int j = 0; j < 4; ++j) {
        int n = c0 + 16 * j + (lane & 15);
        #pragma unroll
        for (int r = 0; r < 8; ++r) {
            int m = q0 + 16 * wave + r + ((lane & 16) ? 8 : 0);
            O[(rowbase + m) * D_MODEL + n] = oacc[j][r] / lrun[r];
        }
    }
}

// ---------------------------------------------------------------------------
// Fused residual + LayerNorm: y = LN(A+B)*g + beta ; one 256-thread block/row.
// ---------------------------------------------------------------------------
__device__ __forceinline__ float block_reduce_sum(float v, float* sm) {
    #pragma unroll
    for (int off = 16; off >= 1; off >>= 1) v += __shfl_xor(v, off, 32);
    int lane = threadIdx.x & 31, wid = threadIdx.x >> 5;
    __syncthreads();
    if (lane == 0) sm[wid] = v;
    __syncthreads();
    float t = sm[0];
    #pragma unroll
    for (int w = 1; w < 8; ++w) t += sm[w];
    return t;
}

__global__ __launch_bounds__(256)
void ln_residual(const float* __restrict__ A, const float* __restrict__ Bsrc,
                 const float* __restrict__ g, const float* __restrict__ be,
                 float* __restrict__ out32, half_t* __restrict__ out16) {
    __shared__ float sm[8];
    const int row = blockIdx.x;
    const int tid = threadIdx.x;
    const float* a = A + (size_t)row * D_MODEL;
    const float* b = Bsrc + (size_t)row * D_MODEL;

    float x[4];
    float s = 0.0f;
    #pragma unroll
    for (int i = 0; i < 4; ++i) {
        int c = tid + i * 256;
        x[i] = a[c] + b[c];
        s += x[i];
    }
    s = block_reduce_sum(s, sm);
    float mu = s * (1.0f / D_MODEL);
    float var = 0.0f;
    #pragma unroll
    for (int i = 0; i < 4; ++i) {
        float d = x[i] - mu;
        var += d * d;
    }
    var = block_reduce_sum(var, sm);
    float rstd = rsqrtf(var * (1.0f / D_MODEL) + LN_EPS);
    #pragma unroll
    for (int i = 0; i < 4; ++i) {
        int c = tid + i * 256;
        float y = (x[i] - mu) * rstd * g[c] + be[c];
        out32[(size_t)row * D_MODEL + c] = y;
        if (out16) out16[(size_t)row * D_MODEL + c] = (half_t)y;
    }
}

// ---------------------------------------------------------------------------
// Host-side orchestration
// ---------------------------------------------------------------------------
extern "C" void kernel_launch(void* const* d_in, const int* in_sizes, int n_in,
                              void* d_out, int out_size, void* d_ws, size_t ws_size,
                              hipStream_t stream) {
    const float* input = (const float*)d_in[0];
    const float* pos   = (const float*)d_in[1];
    const float* Wq    = (const float*)d_in[2];
    const float* Wk    = (const float*)d_in[3];
    const float* Wv    = (const float*)d_in[4];
    const float* W1    = (const float*)d_in[5];
    const float* b1    = (const float*)d_in[6];
    const float* W2    = (const float*)d_in[7];
    const float* b2    = (const float*)d_in[8];
    const float* ln1_g = (const float*)d_in[9];
    const float* ln1_b = (const float*)d_in[10];
    const float* ln2_g = (const float*)d_in[11];
    const float* ln2_b = (const float*)d_in[12];
    float* out = (float*)d_out;

    char* ws = (char*)d_ws;
    size_t off = 0;
    auto alloc = [&](size_t bytes) -> void* {
        void* p = ws + off;
        off += (bytes + 255) & ~(size_t)255;
        return p;
    };

    half_t* qk16  = (half_t*)alloc((size_t)BN * D_MODEL * 2);
    half_t* in16  = (half_t*)alloc((size_t)BN * D_MODEL * 2);
    half_t* Q16   = (half_t*)alloc((size_t)BN * D_MODEL * 2);
    half_t* K16   = (half_t*)alloc((size_t)BN * D_MODEL * 2);
    half_t* V16   = (half_t*)alloc((size_t)BN * D_MODEL * 2);
    half_t* WqT   = (half_t*)alloc((size_t)D_MODEL * D_MODEL * 2);
    half_t* WkT   = (half_t*)alloc((size_t)D_MODEL * D_MODEL * 2);
    half_t* WvT   = (half_t*)alloc((size_t)D_MODEL * D_MODEL * 2);
    half_t* W1T   = (half_t*)alloc((size_t)D_MODEL * FF_DIM * 2);
    half_t* W2T   = (half_t*)alloc((size_t)FF_DIM * D_MODEL * 2);
    half_t* x1_16 = (half_t*)alloc((size_t)BN * D_MODEL * 2);
    half_t* h16   = (half_t*)alloc((size_t)BN * FF_DIM * 2);
    float*  attn32 = (float*)alloc((size_t)BN * D_MODEL * 4);
    float*  x1_32  = (float*)alloc((size_t)BN * D_MODEL * 4);
    float*  ff32   = (float*)alloc((size_t)BN * D_MODEL * 4);

    const int nElem = BN * D_MODEL;
    prep_inputs<<<(nElem + 255) / 256, 256, 0, stream>>>(input, pos, qk16, in16, nElem);

    convert_transpose<<<(D_MODEL * D_MODEL + 255) / 256, 256, 0, stream>>>(Wq, WqT, D_MODEL, D_MODEL);
    convert_transpose<<<(D_MODEL * D_MODEL + 255) / 256, 256, 0, stream>>>(Wk, WkT, D_MODEL, D_MODEL);
    convert_transpose<<<(D_MODEL * D_MODEL + 255) / 256, 256, 0, stream>>>(Wv, WvT, D_MODEL, D_MODEL);
    convert_transpose<<<(D_MODEL * FF_DIM + 255) / 256, 256, 0, stream>>>(W1, W1T, D_MODEL, FF_DIM);
    convert_transpose<<<(FF_DIM * D_MODEL + 255) / 256, 256, 0, stream>>>(W2, W2T, FF_DIM, D_MODEL);

    dim3 blk(128);
    // Q/K/V projections: [4096 x 1024] x [1024 x 1024]
    gemm_wmma<false, false, true, false><<<dim3(D_MODEL / 64, BN / 64), blk, 0, stream>>>(
        qk16, WqT, nullptr, Q16, nullptr, BN, D_MODEL, D_MODEL);
    gemm_wmma<false, false, true, false><<<dim3(D_MODEL / 64, BN / 64), blk, 0, stream>>>(
        qk16, WkT, nullptr, K16, nullptr, BN, D_MODEL, D_MODEL);
    gemm_wmma<false, false, true, false><<<dim3(D_MODEL / 64, BN / 64), blk, 0, stream>>>(
        in16, WvT, nullptr, V16, nullptr, BN, D_MODEL, D_MODEL);

    attn_flash<<<dim3(SEQ / 64, BATCH * NUM_HEADS), blk, 0, stream>>>(Q16, K16, V16, attn32);

    // x = input + attn_out ; x1 = LN1(x)
    ln_residual<<<BN, 256, 0, stream>>>(input, attn32, ln1_g, ln1_b, x1_32, x1_16);

    // h = relu(x1 @ W1 + b1) : [4096 x 1024] x [1024 x 4096]
    gemm_wmma<true, true, true, false><<<dim3(FF_DIM / 64, BN / 64), blk, 0, stream>>>(
        x1_16, W1T, b1, h16, nullptr, BN, FF_DIM, D_MODEL);
    // ff = h @ W2 + b2 : [4096 x 4096] x [4096 x 1024]
    gemm_wmma<true, false, false, true><<<dim3(D_MODEL / 64, BN / 64), blk, 0, stream>>>(
        h16, W2T, b2, nullptr, ff32, BN, D_MODEL, FF_DIM);

    // out = LN2(x1 + ff)
    ln_residual<<<BN, 256, 0, stream>>>(x1_32, ff32, ln2_g, ln2_b, out, nullptr);

    (void)in_sizes; (void)n_in; (void)out_size; (void)ws_size;
}
